// TransformerBlockMultiHeadAttn_88338887344431
// MI455X (gfx1250) — compile-verified, matrix operand order checked by reading, not run on hardware
//
#include <hip/hip_runtime.h>
#include <hip/hip_bf16.h>

// ---------------------------------------------------------------------------
// Problem constants (from reference)
// ---------------------------------------------------------------------------
#define D_MODEL 1024
#define HEADS   16
#define DH      64
#define TSEQ    2048
#define BATCH   2
#define DFF     4096
#define ROWS    (BATCH * TSEQ)   // 4096
#define LN_EPS  1e-5f

typedef __bf16 v16bf __attribute__((ext_vector_type(16)));
typedef float  v8f   __attribute__((ext_vector_type(8)));
typedef unsigned int u32x4 __attribute__((ext_vector_type(4)));
typedef int          i32x8 __attribute__((ext_vector_type(8)));
typedef int          i32x4 __attribute__((ext_vector_type(4)));

union FragU { v16bf v; uint4 q[2]; };

__device__ __forceinline__ unsigned short f2bf(float f) {
    unsigned int u = __float_as_uint(f);
    u += 0x7FFFu + ((u >> 16) & 1u);      // round-to-nearest-even
    return (unsigned short)(u >> 16);
}

// 32-bit LDS byte offset of a __shared__ object (addrspace(3) ptrs are 32-bit)
__device__ __forceinline__ unsigned int lds_byte_addr(const void* p) {
    return (unsigned int)(size_t)(__attribute__((address_space(3))) const void*)p;
}

// Load one 16x32 bf16 WMMA operand fragment (A layout; B identical with
// "row" = output column, data K-contiguous per row).
//   lane l<16 : row l,    K 0..7  (v0-3) and K 16..23 (v4-7)
//   lane l>=16: row l-16, K 8..15 (v0-3) and K 24..31 (v4-7)
__device__ __forceinline__ v16bf load_frag(const unsigned short* base,
                                           int rowStride, int row0, int lane) {
    const unsigned short* r =
        base + (size_t)(row0 + (lane & 15)) * rowStride + ((lane >> 4) << 3);
    FragU f;
    f.q[0] = *(const uint4*)(r);
    f.q[1] = *(const uint4*)(r + 16);
    return f.v;
}

// ---------------------------------------------------------------------------
// Tensor Data Mover: DMA one [128 rows x 64B] bf16 tile global -> LDS.
// 2D descriptor: data_size=4B units, tile_dim0=16 dwords, tile_dim1=128 rows,
// row stride = rowStrideElems/2 dwords.  (CDNA5 ISA)
// This toolchain exposes the 6-arg builtin (g0, g1, g2, g3, extra, cpol).
// ---------------------------------------------------------------------------
__device__ __forceinline__ void tdm_load_tile(unsigned int ldsAddr,
                                              const unsigned short* gptr,
                                              int rowStrideElems) {
    unsigned long long ga = (unsigned long long)(size_t)gptr;
    u32x4 g0;
    g0[0] = 1u;                                            // count=1 (valid)
    g0[1] = ldsAddr;                                       // lds byte addr
    g0[2] = (unsigned int)(ga & 0xFFFFFFFFull);            // global_addr lo
    g0[3] = (unsigned int)((ga >> 32) & 0x1FFFFFFull)      // global_addr hi
            | (2u << 30);                                  // type=2 ("image")
    i32x8 g1;
    g1[0] = (int)(2u << 16);          // data_size=2 -> 4-byte elements
    g1[1] = (int)(16u << 16);         // tensor_dim0 = 16 dwords  (bits 79:48)
    g1[2] = (int)(128u << 16);        // tensor_dim1 = 128 rows   (bits 111:80)
    g1[3] = (int)(16u << 16);         // tile_dim0   = 16 dwords  (bits 127:112)
    g1[4] = 128;                      // tile_dim1   = 128 rows   (bits 143:128)
    g1[5] = rowStrideElems >> 1;      // tensor_dim0_stride (dwords, bits 207:160)
    g1[6] = 0;
    g1[7] = 0;
    i32x4 z4 = {};
    i32x8 z8 = {};
    __builtin_amdgcn_tensor_load_to_lds(g0, g1, z4, z4, z8, 0);
}

// ---------------------------------------------------------------------------
// All-bf16 WMMA GEMM with TDM double-buffered staging:
//   C[M,N] = act( A[M,K](bf16) @ Bt[N,K](bf16)^T + bias[N] )
// Block = 256 threads (8 waves), tile 128x128, BK=32; wave 0 drives the DMA.
// ---------------------------------------------------------------------------
template <bool OUT_BF16, bool RELU>
__global__ __launch_bounds__(256) void gemm_bb_kernel(
    const unsigned short* __restrict__ A,   // [M][K] bf16 row-major
    const unsigned short* __restrict__ Bt,  // [N][K] bf16 row-major (W^T)
    const float* __restrict__ bias, void* __restrict__ Cout,
    int M, int N, int K) {
    __shared__ unsigned short sA[2][128 * 32];
    __shared__ unsigned short sB[2][128 * 32];

    const int tid  = threadIdx.x;
    const int lane = tid & 31;
    const int w    = tid >> 5;    // 0..7
    const int wr   = w >> 1;      // 0..3  -> 32-row slab
    const int wc   = w & 1;       // 0..1  -> 64-col slab
    const int bm   = blockIdx.y * 128;
    const int bn   = blockIdx.x * 128;
    const int hi   = lane >> 4;
    const int nk   = K >> 5;

    v8f acc[2][4];
    for (int i = 0; i < 2; ++i)
        for (int j = 0; j < 4; ++j) acc[i][j] = v8f{};

    // prologue: wave 0 DMAs tile 0 into buffer 0
    if (w == 0) {
        tdm_load_tile(lds_byte_addr(&sA[0][0]), A + (size_t)bm * K, K);
        tdm_load_tile(lds_byte_addr(&sB[0][0]), Bt + (size_t)bn * K, K);
    }

    for (int kt = 0; kt < nk; ++kt) {
        const int cur = kt & 1, nxt = cur ^ 1;
        if (w == 0) {
            if (kt + 1 < nk) {
                const int k1 = (kt + 1) << 5;
                tdm_load_tile(lds_byte_addr(&sA[nxt][0]),
                              A + (size_t)bm * K + k1, K);
                tdm_load_tile(lds_byte_addr(&sB[nxt][0]),
                              Bt + (size_t)bn * K + k1, K);
                // keep only the 2 just-issued DMAs outstanding:
                // the pair for tile kt has landed.
                __builtin_amdgcn_s_wait_tensorcnt(2);
            } else {
                __builtin_amdgcn_s_wait_tensorcnt(0);
            }
        }
        __syncthreads();   // tile kt visible to all waves

        v16bf af[2], bfr[4];
        for (int ms = 0; ms < 2; ++ms)
            af[ms] = load_frag(sA[cur], 32, wr * 32 + ms * 16, lane);
        for (int ns = 0; ns < 4; ++ns)
            bfr[ns] = load_frag(sB[cur], 32, wc * 64 + ns * 16, lane);
        for (int ms = 0; ms < 2; ++ms)
            for (int ns = 0; ns < 4; ++ns)
                acc[ms][ns] = __builtin_amdgcn_wmma_f32_16x16x32_bf16(
                    false, af[ms], false, bfr[ns], (short)0, acc[ms][ns],
                    false, false);

        __syncthreads();   // done reading buf 'cur' before it is re-filled
    }

    // Epilogue: C layout — lane holds col n=(lane&15); rows m = v + 8*hi.
    for (int ms = 0; ms < 2; ++ms)
        for (int ns = 0; ns < 4; ++ns) {
            int n = bn + wc * 64 + ns * 16 + (lane & 15);
            float bv = bias[n];
            for (int v = 0; v < 8; ++v) {
                int m = bm + wr * 32 + ms * 16 + v + hi * 8;
                float val = acc[ms][ns][v] + bv;
                if (RELU) val = fmaxf(val, 0.f);
                if (OUT_BF16)
                    ((unsigned short*)Cout)[(size_t)m * N + n] = f2bf(val);
                else
                    ((float*)Cout)[(size_t)m * N + n] = val;
            }
        }
}

// ---------------------------------------------------------------------------
// Attention: flash-style online softmax, WMMA for QK^T and P@V.
// Block = 128 threads (4 waves); wave w handles 16 queries. grid=(T/64,H,B).
// Q/K/V bf16 [ROWS][D_MODEL]; output bf16 (feeds the Wo GEMM via TDM).
// ---------------------------------------------------------------------------
__global__ __launch_bounds__(128) void attn_kernel(
    const unsigned short* __restrict__ Qb, const unsigned short* __restrict__ Kb,
    const unsigned short* __restrict__ Vb, const float* __restrict__ dscale_p,
    unsigned short* __restrict__ Aout) {
    __shared__ unsigned short sP[4][16 * 32];   // per-wave P tile  [m][k]
    __shared__ unsigned short sVt[4][64 * 32];  // per-wave V^T     [d][k]

    const int tid  = threadIdx.x;
    const int lane = tid & 31;
    const int w    = tid >> 5;
    const int hi   = lane >> 4;
    const int h    = blockIdx.y;
    const int b    = blockIdx.z;
    const int q0   = blockIdx.x * 64 + w * 16;
    const int rowbase = b * TSEQ;
    const float dscale = *dscale_p;
    const float inv_sqrt_dh = 0.125f;   // 1/sqrt(64)

    v16bf qf[2];
    for (int c = 0; c < 2; ++c)
        qf[c] = load_frag(Qb + (size_t)(rowbase + q0) * D_MODEL + h * DH + c * 32,
                          D_MODEL, 0, lane);

    float mrow[8], lrow[8], alpha[8];
    v8f o[4];
    for (int v = 0; v < 8; ++v) { mrow[v] = -INFINITY; lrow[v] = 0.f; }
    for (int t = 0; t < 4; ++t) o[t] = v8f{};

    for (int u0 = 0; u0 < TSEQ; u0 += 32) {
        // ---- S = Q @ K^T : two 16-key tiles --------------------------------
        v8f s[2];
        for (int t = 0; t < 2; ++t) {
            s[t] = v8f{};
            for (int c = 0; c < 2; ++c) {
                v16bf kf = load_frag(
                    Kb + (size_t)(rowbase + u0 + t * 16) * D_MODEL + h * DH + c * 32,
                    D_MODEL, 0, lane);
                s[t] = __builtin_amdgcn_wmma_f32_16x16x32_bf16(
                    false, qf[c], false, kf, (short)0, s[t], false, false);
            }
        }
        // ---- distance bias, faithful (u,h) mask, scale ---------------------
        for (int t = 0; t < 2; ++t) {
            int u = u0 + t * 16 + (lane & 15);
            for (int v = 0; v < 8; ++v) {
                int q = q0 + v + hi * 8;
                float sv = s[t][v] - dscale * fabsf((float)(q - u));
                if (h > u) sv = -INFINITY;   // mask over (key u, head h)
                s[t][v] = sv * inv_sqrt_dh;
            }
        }
        // ---- online softmax update ----------------------------------------
        for (int v = 0; v < 8; ++v) {
            float mx = fmaxf(s[0][v], s[1][v]);
            for (int m = 1; m < 16; m <<= 1)
                mx = fmaxf(mx, __shfl_xor(mx, m, 32));   // within 16-lane half
            float mnew = fmaxf(mrow[v], mx);
            float a  = __expf(mrow[v] - mnew);
            float p0 = __expf(s[0][v] - mnew);
            float p1 = __expf(s[1][v] - mnew);
            s[0][v] = p0; s[1][v] = p1;
            float rs = p0 + p1;
            for (int m = 1; m < 16; m <<= 1) rs += __shfl_xor(rs, m, 32);
            lrow[v] = lrow[v] * a + rs;
            mrow[v] = mnew;
            alpha[v] = a;
        }
        for (int t = 0; t < 4; ++t)
            for (int v = 0; v < 8; ++v) o[t][v] *= alpha[v];

        // ---- stage P (bf16, A-operand layout) and V^T in per-wave LDS ------
        for (int t = 0; t < 2; ++t)
            for (int v = 0; v < 8; ++v)
                sP[w][(v + hi * 8) * 32 + t * 16 + (lane & 15)] = f2bf(s[t][v]);
        {
            const unsigned short* vrow =
                Vb + (size_t)(rowbase + u0 + lane) * D_MODEL + h * DH;
            for (int d = 0; d < 64; ++d)
                sVt[w][d * 32 + lane] = vrow[d];
        }
        asm volatile("s_wait_dscnt 0" ::: "memory");  // cross-lane LDS dep

        // ---- O += P @ V ----------------------------------------------------
        v16bf pf = load_frag(sP[w], 32, 0, lane);
        for (int t = 0; t < 4; ++t) {
            v16bf vf = load_frag(sVt[w], 32, t * 16, lane);
            o[t] = __builtin_amdgcn_wmma_f32_16x16x32_bf16(
                false, pf, false, vf, (short)0, o[t], false, false);
        }
    }

    // ---- normalize and store bf16 (B,T,H,dh) flattened to D ----------------
    for (int t = 0; t < 4; ++t) {
        int n = h * DH + t * 16 + (lane & 15);
        for (int v = 0; v < 8; ++v) {
            int q = q0 + v + hi * 8;
            Aout[(size_t)(rowbase + q) * D_MODEL + n] = f2bf(o[t][v] / lrow[v]);
        }
    }
}

// ---------------------------------------------------------------------------
// out = LayerNorm(a + c) * g + beta ; optional bf16 shadow copy for next GEMM
// ---------------------------------------------------------------------------
template <bool EMIT_BF16>
__global__ __launch_bounds__(256) void add_ln_kernel(
    const float* __restrict__ a, const float* __restrict__ c,
    const float* __restrict__ g, const float* __restrict__ beta,
    float* __restrict__ out, unsigned short* __restrict__ outb) {
    __shared__ float red[8];
    const int row = blockIdx.x;
    const int tid = threadIdx.x;
    const float* pa = a + (size_t)row * D_MODEL;
    const float* pc = c + (size_t)row * D_MODEL;

    float x[4];
    float s = 0.f;
    for (int i = 0; i < 4; ++i) {
        x[i] = pa[tid + i * 256] + pc[tid + i * 256];
        s += x[i];
    }
    for (int m = 1; m < 32; m <<= 1) s += __shfl_xor(s, m, 32);
    if ((tid & 31) == 0) red[tid >> 5] = s;
    __syncthreads();
    float tot = 0.f;
    for (int i = 0; i < 8; ++i) tot += red[i];
    const float mu = tot / (float)D_MODEL;
    __syncthreads();

    float vs = 0.f;
    for (int i = 0; i < 4; ++i) { float d = x[i] - mu; vs += d * d; }
    for (int m = 1; m < 32; m <<= 1) vs += __shfl_xor(vs, m, 32);
    if ((tid & 31) == 0) red[tid >> 5] = vs;
    __syncthreads();
    float vtot = 0.f;
    for (int i = 0; i < 8; ++i) vtot += red[i];
    const float rstd = rsqrtf(vtot / (float)D_MODEL + LN_EPS);

    for (int i = 0; i < 4; ++i) {
        int cc = tid + i * 256;
        float val = (x[i] - mu) * rstd * g[cc] + beta[cc];
        out[(size_t)row * D_MODEL + cc] = val;
        if (EMIT_BF16) outb[(size_t)row * D_MODEL + cc] = f2bf(val);
    }
}

// ---------------------------------------------------------------------------
// Prologue converters (run once; amortized over 32 M-tiles of GEMM reuse)
// ---------------------------------------------------------------------------
__global__ __launch_bounds__(256) void cvt_bf16_kernel(
    const float* __restrict__ in, unsigned short* __restrict__ out) {
    int base = blockIdx.x * 2048 + threadIdx.x;
    for (int j = 0; j < 8; ++j) out[base + j * 256] = f2bf(in[base + j * 256]);
}

// W[K][N] f32 -> Wt[N][K] bf16  (LDS-tiled transpose), block (32,8)
__global__ void wtrans_kernel(const float* __restrict__ W,
                              unsigned short* __restrict__ Wt, int K, int N) {
    __shared__ float tile[32][33];
    const int n0 = blockIdx.x * 32, k0 = blockIdx.y * 32;
    const int tx = threadIdx.x, ty = threadIdx.y;
    for (int i = 0; i < 4; ++i)
        tile[ty + 8 * i][tx] = W[(size_t)(k0 + ty + 8 * i) * N + (n0 + tx)];
    __syncthreads();
    for (int i = 0; i < 4; ++i)
        Wt[(size_t)(n0 + ty + 8 * i) * K + (k0 + tx)] = f2bf(tile[tx][ty + 8 * i]);
}

// ---------------------------------------------------------------------------
// Host-side orchestration
// ---------------------------------------------------------------------------
extern "C" void kernel_launch(void* const* d_in, const int* in_sizes, int n_in,
                              void* d_out, int out_size, void* d_ws,
                              size_t ws_size, hipStream_t stream) {
    const float* tokens = (const float*)d_in[0];
    const float* Wq = (const float*)d_in[1];
    const float* bq = (const float*)d_in[2];
    const float* Wk = (const float*)d_in[3];
    const float* bk = (const float*)d_in[4];
    const float* Wv = (const float*)d_in[5];
    const float* bv = (const float*)d_in[6];
    const float* Wo = (const float*)d_in[7];
    const float* bo = (const float*)d_in[8];
    const float* dist_scale = (const float*)d_in[9];
    const float* W1 = (const float*)d_in[10];
    const float* b1 = (const float*)d_in[11];
    const float* W2 = (const float*)d_in[12];
    const float* b2 = (const float*)d_in[13];
    const float* ln1_g = (const float*)d_in[14];
    const float* ln1_b = (const float*)d_in[15];
    const float* ln2_g = (const float*)d_in[16];
    const float* ln2_b = (const float*)d_in[17];

    char* ws = (char*)d_ws;
    const size_t MB = 1ull << 20;
    unsigned short* tokensb = (unsigned short*)(ws + 0 * MB);    // 8 MB bf16
    unsigned short* attnOb  = (unsigned short*)(ws + 0 * MB);    // reuses tokensb
    unsigned short* Qb      = (unsigned short*)(ws + 8 * MB);    // 8 MB
    unsigned short* Kb      = (unsigned short*)(ws + 16 * MB);   // 8 MB
    unsigned short* Vb      = (unsigned short*)(ws + 24 * MB);   // 8 MB
    float*          attnP   = (float*)(ws + 32 * MB);            // 16 MB f32
    float*          xbuf    = (float*)(ws + 48 * MB);            // 16 MB f32
    unsigned short* xb      = (unsigned short*)(ws + 64 * MB);   // 8 MB bf16
    unsigned short* hffn    = (unsigned short*)(ws + 72 * MB);   // 32 MB bf16
    float*          ybuf    = (float*)(ws + 104 * MB);           // 16 MB f32
    unsigned short* Wqt     = (unsigned short*)(ws + 120 * MB);  // 2 MB each
    unsigned short* Wkt     = (unsigned short*)(ws + 122 * MB);
    unsigned short* Wvt     = (unsigned short*)(ws + 124 * MB);
    unsigned short* Wot     = (unsigned short*)(ws + 126 * MB);
    unsigned short* W1t     = (unsigned short*)(ws + 128 * MB);  // 8 MB [4096][1024]
    unsigned short* W2t     = (unsigned short*)(ws + 136 * MB);  // 8 MB [1024][4096]

    // --- one-time bf16 conversion / weight transposition --------------------
    cvt_bf16_kernel<<<(ROWS * D_MODEL) / 2048, 256, 0, stream>>>(tokens, tokensb);
    wtrans_kernel<<<dim3(32, 32),  dim3(32, 8), 0, stream>>>(Wq, Wqt, D_MODEL, D_MODEL);
    wtrans_kernel<<<dim3(32, 32),  dim3(32, 8), 0, stream>>>(Wk, Wkt, D_MODEL, D_MODEL);
    wtrans_kernel<<<dim3(32, 32),  dim3(32, 8), 0, stream>>>(Wv, Wvt, D_MODEL, D_MODEL);
    wtrans_kernel<<<dim3(32, 32),  dim3(32, 8), 0, stream>>>(Wo, Wot, D_MODEL, D_MODEL);
    wtrans_kernel<<<dim3(128, 32), dim3(32, 8), 0, stream>>>(W1, W1t, D_MODEL, DFF);
    wtrans_kernel<<<dim3(32, 128), dim3(32, 8), 0, stream>>>(W2, W2t, DFF, D_MODEL);

    // --- Q/K/V projections (TDM-staged WMMA GEMMs, bf16 out) ---------------
    gemm_bb_kernel<true, false><<<dim3(8, 32), 256, 0, stream>>>(
        tokensb, Wqt, bq, Qb, ROWS, D_MODEL, D_MODEL);
    gemm_bb_kernel<true, false><<<dim3(8, 32), 256, 0, stream>>>(
        tokensb, Wkt, bk, Kb, ROWS, D_MODEL, D_MODEL);
    gemm_bb_kernel<true, false><<<dim3(8, 32), 256, 0, stream>>>(
        tokensb, Wvt, bv, Vb, ROWS, D_MODEL, D_MODEL);

    // --- attention (writes bf16; tokensb region is dead by now) ------------
    attn_kernel<<<dim3(TSEQ / 64, HEADS, BATCH), 128, 0, stream>>>(
        Qb, Kb, Vb, dist_scale, attnOb);

    // --- output projection --------------------------------------------------
    gemm_bb_kernel<false, false><<<dim3(8, 32), 256, 0, stream>>>(
        attnOb, Wot, bo, attnP, ROWS, D_MODEL, D_MODEL);

    // --- x = LN(tokens + attn), with bf16 shadow for FFN input -------------
    add_ln_kernel<true><<<ROWS, 256, 0, stream>>>(tokens, attnP, ln1_g, ln1_b,
                                                  xbuf, xb);

    // --- FFN ----------------------------------------------------------------
    gemm_bb_kernel<true, true><<<dim3(32, 32), 256, 0, stream>>>(
        xb, W1t, b1, hffn, ROWS, DFF, D_MODEL);
    gemm_bb_kernel<false, false><<<dim3(8, 32), 256, 0, stream>>>(
        hffn, W2t, b2, ybuf, ROWS, D_MODEL, DFF);

    // --- z = LN(y + x) ------------------------------------------------------
    add_ln_kernel<false><<<ROWS, 256, 0, stream>>>(ybuf, xbuf, ln2_g, ln2_b,
                                                   (float*)d_out, nullptr);
}
